// RNN_80762565034141
// MI455X (gfx1250) — compile-verified, matrix operand order checked by reading, not run on hardware
//
#include <hip/hip_runtime.h>
#include <hip/hip_fp16.h>
#include <hip/hip_bf16.h>

typedef __attribute__((ext_vector_type(16))) _Float16 v16h;
typedef __attribute__((ext_vector_type(8)))  float    v8f;
typedef unsigned int u32x4 __attribute__((ext_vector_type(4)));
typedef int          i32x4 __attribute__((ext_vector_type(4)));
typedef int          i32x8 __attribute__((ext_vector_type(8)));

#define NN 2048
#define BB 32
#define TT 1024
#define TRR 256
#define NT_BLOCKS 16          // workgroups in persistent kernel
#define WAVES_PER_BLOCK 8     // 256 threads, wave32
#define DT 0.1f
#define ONE_MINUS_DT 0.9f
#define NOISE_SCALE 0.0158113883f   // sqrt(0.1)*0.05
#define KT_TILES 64           // 2048 / 32
#define TILE_ELEMS 512        // 32x16 f16 B-operand tile

// ---------------------------------------------------------------------------
// Prep: W_eff = |wrec| * mask, transposed to [k][n] and swizzled into the
// per-lane WMMA B-operand tile layout: tile (ntg, kt) is 512 contiguous f16,
// lane L holds 16 contiguous elements (col = L&15, K-half = L>>4).
// Also zeroes the per-step barrier counters (re-run every launch => graph-safe).
// ---------------------------------------------------------------------------
__global__ void rnn_prep_w(const float* __restrict__ wrec,
                           const float* __restrict__ mask,
                           _Float16* __restrict__ wt,
                           unsigned* __restrict__ counters) {
  const int idx = blockIdx.x * blockDim.x + threadIdx.x;   // 0 .. N*N-1
  const int e    = idx & 511;         // element within 32x16 tile
  const int kt   = (idx >> 9) & 63;   // K tile
  const int ntg  = idx >> 15;         // 16-col tile (0..127)
  const int lane = e >> 4;
  const int i    = e & 15;
  const int nn   = lane & 15;
  const int kk   = ((lane >> 4) << 4) + i;
  const int k = kt * 32 + kk;
  const int n = ntg * 16 + nn;
  const size_t src = (size_t)n * NN + k;          // W_eff[n][k] -> B[k][n]
  wt[idx] = (_Float16)(fabsf(wrec[src]) * mask[src]);
  if (idx < TT) counters[idx] = 0u;
}

// s0 = sigmoid(h_init) in f16 (A-operand buffer for step 0)
__global__ void rnn_init_s(const float* __restrict__ h_init,
                           _Float16* __restrict__ s0) {
  const int idx = blockIdx.x * blockDim.x + threadIdx.x;   // 0 .. B*N-1
  const float h = h_init[idx];
  s0[idx] = (_Float16)(1.0f / (1.0f + __expf(-h)));
}

// ---------------------------------------------------------------------------
// Persistent RNN kernel. grid = 16 WGs x 256 thr. Each wave owns a 32x16
// (batch x neuron) tile: h lives in registers for all 1024 steps; only the
// 128KB sigmoid(h) vector round-trips through an L2-resident double buffer
// guarded by a per-step device-scope counter barrier. Per-step s staging is a
// single Tensor Data Mover DMA (tensor_load_to_lds) per workgroup.
// ---------------------------------------------------------------------------
__global__ void __launch_bounds__(256, 1)
rnn_persistent(const float* __restrict__ input,
               const float* __restrict__ h_init,
               const float* __restrict__ noise,
               const _Float16* __restrict__ wt,
               _Float16* __restrict__ sbuf0,
               _Float16* __restrict__ sbuf1,
               unsigned* __restrict__ counters,
               float* __restrict__ out) {
  extern __shared__ _Float16 s_lds[];            // 32 x 2048 f16 = 128 KB

  const int tid   = threadIdx.x;
  const int lane  = tid & 31;
  const int wave  = tid >> 5;
  const int ntg   = blockIdx.x * WAVES_PER_BLOCK + wave;   // 0..127
  const int n     = ntg * 16 + (lane & 15);
  const int khalf = lane >> 4;
  const int rbase = khalf * 8;                   // C-layout: rows r + 8*(lane>>4)
  const _Float16* __restrict__ wt_wave = wt + (size_t)ntg * (KT_TILES * TILE_ELEMS);

  // h state for this wave's 32x16 tile stays in VGPRs across all steps
  float h0[8], h1[8];
#pragma unroll
  for (int r = 0; r < 8; ++r) {
    h0[r] = h_init[(size_t)(rbase + r) * NN + n];
    h1[r] = h_init[(size_t)(16 + rbase + r) * NN + n];
  }

  _Float16* sb[2] = { sbuf0, sbuf1 };

#pragma unroll 1
  for (int t = 0; t < TT; ++t) {
    const _Float16* __restrict__ s_cur = sb[t & 1];
    _Float16* __restrict__ s_nxt = sb[(t + 1) & 1];

    // ---- wait until every WG finished step t-1 (s_cur fully published) ----
    if (t > 0) {
      if (tid == 0) {
        while (__hip_atomic_load(&counters[t - 1], __ATOMIC_ACQUIRE,
                                 __HIP_MEMORY_SCOPE_AGENT) < (unsigned)NT_BLOCKS) {
          __builtin_amdgcn_s_sleep(2);
        }
      }
      __syncthreads();
    }

    // ---- stage s (32x2048 f16 = 128 KB) into LDS via the Tensor Data Mover --
#if __has_builtin(__builtin_amdgcn_tensor_load_to_lds)
    if (wave == 0) {
      const unsigned lds_base = (unsigned)(size_t)(void*)s_lds;
      const unsigned long long ga = (unsigned long long)(size_t)s_cur;
      // D# group 0: count=1 | lds_addr | global_addr[56:0] | type=2
      u32x4 g0;
      g0[0] = 1u;
      g0[1] = lds_base;
      g0[2] = (unsigned)(ga & 0xffffffffu);
      g0[3] = (unsigned)((ga >> 32) & 0x01ffffffu) | (2u << 30);
      // D# group 1: data_size=8B; 1-D tile of 16384 units (=128 KB)
      i32x8 g1;
      g1[0] = 0x00030000;            // workgroup_mask=0, data_size=3 (8B)
      g1[1] = (int)0x40000000;       // tensor_dim0[15:0]=16384 in bits [63:48]
      g1[2] = 0x00010000;            // tensor_dim0[31:16]=0, tensor_dim1=1
      g1[3] = (int)0x40000000;       // tile_dim0=16384 in bits [127:112]
      g1[4] = 0;                     // tile_dim1=0 (unused), tile_dim2=0
      g1[5] = 16384;                 // tensor_dim0_stride[31:0]
      g1[6] = 0;
      g1[7] = 0;
      i32x4 gz4 = {0, 0, 0, 0};      // groups 2/3: tensor <= 2D
      i32x8 gz8 = {0, 0, 0, 0, 0, 0, 0, 0};
      __builtin_amdgcn_tensor_load_to_lds(g0, g1, gz4, gz4, gz8, 0);
      __builtin_amdgcn_s_wait_tensorcnt(0);
    }
#else
    {
      const uint4* __restrict__ src = (const uint4*)s_cur;
      uint4* dst = (uint4*)s_lds;
#pragma unroll 4
      for (int i = tid; i < (BB * NN * 2) / 16; i += 256) dst[i] = src[i];
    }
#endif
    __syncthreads();

    // ---- rec_in = s @ W_eff^T for this wave's 32x16 tile --------------------
    v8f acc0 = {0.f, 0.f, 0.f, 0.f, 0.f, 0.f, 0.f, 0.f};
    v8f acc1 = {0.f, 0.f, 0.f, 0.f, 0.f, 0.f, 0.f, 0.f};
    const int arow0 = (lane & 15) * NN + khalf * 16;
    const int arow1 = arow0 + 16 * NN;
#pragma unroll 4
    for (int kt = 0; kt < KT_TILES; ++kt) {
      __builtin_prefetch(wt_wave + (size_t)(kt + 8) * TILE_ELEMS + lane * 16, 0, 1);
      v16h a0 = *(const v16h*)(s_lds + arow0 + kt * 32);
      v16h a1 = *(const v16h*)(s_lds + arow1 + kt * 32);
      v16h bt = *(const v16h*)(wt_wave + (size_t)kt * TILE_ELEMS + lane * 16);
      acc0 = __builtin_amdgcn_wmma_f32_16x16x32_f16(false, a0, false, bt,
                                                    (short)0, acc0, false, false);
      acc1 = __builtin_amdgcn_wmma_f32_16x16x32_f16(false, a1, false, bt,
                                                    (short)0, acc1, false, false);
    }

    // ---- fused leaky update + noise + sigmoid + publish --------------------
    const int  tr  = t >> 2;
    const bool rec = ((t & 3) == 3);
#pragma unroll
    for (int r = 0; r < 8; ++r) {
      {
        const int b = rbase + r;
        const size_t gi = ((size_t)b * TT + t) * NN + n;
        const float hr = ONE_MINUS_DT * h0[r] + DT * (acc0[r] + input[gi])
                         + NOISE_SCALE * noise[gi];
        h0[r] = hr;
        const float s = 1.0f / (1.0f + __expf(-hr));
        s_nxt[(size_t)b * NN + n] = (_Float16)s;
        if (rec) out[((size_t)b * TRR + tr) * NN + n] = s;
      }
      {
        const int b = 16 + rbase + r;
        const size_t gi = ((size_t)b * TT + t) * NN + n;
        const float hr = ONE_MINUS_DT * h1[r] + DT * (acc1[r] + input[gi])
                         + NOISE_SCALE * noise[gi];
        h1[r] = hr;
        const float s = 1.0f / (1.0f + __expf(-hr));
        s_nxt[(size_t)b * NN + n] = (_Float16)s;
        if (rec) out[((size_t)b * TRR + tr) * NN + n] = s;
      }
    }

    // ---- arrive: all our s_nxt stores visible device-wide ------------------
    __threadfence();
    __syncthreads();
    if (tid == 0) {
      __hip_atomic_fetch_add(&counters[t], 1u, __ATOMIC_RELEASE,
                             __HIP_MEMORY_SCOPE_AGENT);
    }
  }
}

// ---------------------------------------------------------------------------
extern "C" void kernel_launch(void* const* d_in, const int* in_sizes, int n_in,
                              void* d_out, int out_size, void* d_ws, size_t ws_size,
                              hipStream_t stream) {
  const float* input  = (const float*)d_in[0];   // (B,T,N)
  const float* h_init = (const float*)d_in[1];   // (B,N)
  const float* noise  = (const float*)d_in[2];   // (B,T,N)
  const float* wrec   = (const float*)d_in[3];   // (N,N)
  const float* mask   = (const float*)d_in[4];   // (N,N)
  float* out = (float*)d_out;                    // (B,Tr,N)

  char* ws = (char*)d_ws;
  _Float16* wt = (_Float16*)ws;                               // 8 MB tiled f16 W
  _Float16* s0 = (_Float16*)(ws + (size_t)NN * NN * 2);       // 128 KB
  _Float16* s1 = s0 + (size_t)BB * NN;                        // 128 KB
  unsigned* counters = (unsigned*)(ws + (size_t)NN * NN * 2
                                      + 2 * (size_t)BB * NN * 2);

  rnn_prep_w<<<(NN * NN) / 256, 256, 0, stream>>>(wrec, mask, wt, counters);
  rnn_init_s<<<(BB * NN) / 256, 256, 0, stream>>>(h_init, s0);
  rnn_persistent<<<NT_BLOCKS, 256, BB * NN * sizeof(_Float16), stream>>>(
      input, h_init, noise, wt, s0, s1, counters, out);
}
